// GraphSAGECluster_5282809774427
// MI455X (gfx1250) — matrix operand order repaired and checked by reading, not compiled, not run on hardware
//
#include <hip/hip_runtime.h>

// ---------------------------------------------------------------------------
// GraphSAGECluster fused pipeline for gfx1250 (MI455X, wave32, WMMA).
// GEMMs run on v_wmma_f32_16x16x32_bf16 (bf16 in, f32 accumulate).
// ---------------------------------------------------------------------------

namespace {

constexpr int NN = 6144;    // nodes
constexpr int EE = 196608;  // edges per graph
constexpr int FF = 256;     // input feature dim
constexpr int HH = 256;     // hidden dim
constexpr int DD = 512;     // fusion dim
constexpr int GG = 4;       // graphs
constexpr int AT = 128;     // attention bottleneck

typedef __attribute__((ext_vector_type(16))) __bf16 v16bf;
typedef __attribute__((ext_vector_type(8)))  float  v8f;

__device__ __forceinline__ __bf16 f2bf(float f) { return (__bf16)f; }

// ---------------------------- elementwise helpers --------------------------

__global__ __launch_bounds__(256) void fill_f32(float* __restrict__ p, float v, int n4) {
  int i = blockIdx.x * 256 + threadIdx.x;
  if (i < n4) ((float4*)p)[i] = make_float4(v, v, v, v);
}

__global__ __launch_bounds__(256) void copy_f32(float* __restrict__ dst,
                                                const float* __restrict__ src, int n4) {
  int i = blockIdx.x * 256 + threadIdx.x;
  if (i < n4) ((float4*)dst)[i] = ((const float4*)src)[i];
}

// ----------------------------- degree handling -----------------------------

__global__ __launch_bounds__(256) void count_deg(const int* __restrict__ ei,
                                                 float* __restrict__ deg) {
  int g = blockIdx.z;
  int e = blockIdx.x * 256 + threadIdx.x;
  if (e < EE) {
    int dst = ei[(size_t)g * 2 * EE + EE + e];
    atomicAdd(deg + (size_t)g * NN + dst, 1.0f);
  }
}

__global__ __launch_bounds__(256) void inv_deg(float* __restrict__ deg, int n) {
  int i = blockIdx.x * 256 + threadIdx.x;
  if (i < n) deg[i] = 1.0f / fmaxf(deg[i], 1.0f);
}

// --------------------- edge scatter (segment_sum of x[src]) ----------------
// block = 256 threads = 4 edges x 64 threads; each thread moves float4.

__global__ __launch_bounds__(256) void scatter_sum(const float* __restrict__ x,
                                                   const int* __restrict__ ei,
                                                   float* __restrict__ m) {
  int g = blockIdx.z;
  int e = blockIdx.x * 4 + (threadIdx.x >> 6);
  int f = (threadIdx.x & 63) << 2;
  int src = ei[(size_t)g * 2 * EE + e];
  int dst = ei[(size_t)g * 2 * EE + EE + e];
  const float4 v = *(const float4*)(x + (size_t)g * NN * HH + (size_t)src * HH + f);
  float* mp = m + (size_t)g * NN * HH + (size_t)dst * HH + f;
  atomicAdd(mp + 0, v.x);
  atomicAdd(mp + 1, v.y);
  atomicAdd(mp + 2, v.z);
  atomicAdd(mp + 3, v.w);
}

// ----------------------------- WMMA GEMM -----------------------------------
// Out[g][row][col] = act( A1*rs1 @ W1^T + (A2 @ W2^T) + bias )
// A: [M,K] f32 row-major. W: [Nout,K] f32 row-major (as in torch lin.weight).
// One wave -> 16x64 output tile (4 accumulators). 8 waves / block.
// All branches around WMMA are wave-uniform (EXEC stays all-ones).

__global__ __launch_bounds__(256)
void wmma_gemm(const float* __restrict__ A1, const float* __restrict__ rs1,
               const float* __restrict__ W1,
               const float* __restrict__ A2, const float* __restrict__ W2,
               const float* __restrict__ bias, float* __restrict__ Out,
               int M, int K, int Nout,
               long long a_gs, long long w_gs, long long b_gs, long long o_gs,
               long long rs_gs, int relu)
{
  const int g = blockIdx.z;
  A1 += (long long)g * a_gs;
  W1 += (long long)g * w_gs;
  if (A2)   A2   += (long long)g * a_gs;
  if (W2)   W2   += (long long)g * w_gs;
  if (bias) bias += (long long)g * b_gs;
  if (rs1)  rs1  += (long long)g * rs_gs;
  Out += (long long)g * o_gs;

  const int lane = threadIdx.x & 31;
  const int wave = threadIdx.x >> 5;
  const int half = lane >> 4;
  const int r16  = lane & 15;

  const int ctiles = Nout >> 6;              // 64-wide column tiles
  const int tile   = blockIdx.x * 8 + wave;  // wave-uniform
  const int ntiles = (M >> 4) * ctiles;
  if (tile >= ntiles) return;
  const int mtile = tile / ctiles;
  const int ctile = tile - mtile * ctiles;
  const int row0  = mtile << 4;
  const int col0  = ctile << 6;

  v8f acc[4] = {};

  const int arow = row0 + r16;  // A-fragment row for this lane

  for (int pass = 0; pass < 2; ++pass) {
    const float* A = pass ? A2 : A1;
    const float* W = pass ? W2 : W1;
    if (!A) continue;  // wave-uniform
    const float rs = (pass == 0 && rs1) ? rs1[arow] : 1.0f;
    const float* arowp = A + (long long)arow * K;

    for (int k0 = 0; k0 < K; k0 += 32) {
      __builtin_prefetch(arowp + k0 + 32, 0, 1);  // global_prefetch_b8
      // A fragment: 16x32 bf16, element i holds K (k0 + 16*half + i)
      v16bf a;
#pragma unroll
      for (int i = 0; i < 16; ++i) {
        const int kk = 16 * half + i;
        a[i] = f2bf(arowp[k0 + kk] * rs);
      }
#pragma unroll
      for (int c = 0; c < 4; ++c) {
        const int col = col0 + (c << 4) + r16;
        const float* wp = W + (long long)col * K + k0 + 16 * half;
        // B fragment: 32x16 bf16, element i holds K row (k0 + 16*half + i)
        v16bf b;
#pragma unroll
        for (int i = 0; i < 16; ++i) b[i] = f2bf(wp[i]);
        acc[c] = __builtin_amdgcn_wmma_f32_16x16x32_bf16(
            false, a, false, b, (short)0, acc[c], false, false);
      }
    }
  }

#pragma unroll
  for (int c = 0; c < 4; ++c) {
    const int col = col0 + (c << 4) + r16;
    const float bv = bias ? bias[col] : 0.0f;
#pragma unroll
    for (int r = 0; r < 8; ++r) {
      const int row = row0 + r + 8 * half;  // C/D layout: lanes 16-31 -> M+8
      float v = acc[c][r] + bv;
      if (relu) v = fmaxf(v, 0.0f);
      Out[(long long)row * Nout + col] = v;
    }
  }
}

// ------------------------- attention logits (GEMV) -------------------------

__global__ __launch_bounds__(256) void logits_kernel(const float* __restrict__ h,
                                                     const float* __restrict__ W2,
                                                     const float* __restrict__ b2,
                                                     float* __restrict__ logits) {
  int i = blockIdx.x * 256 + threadIdx.x;  // over G*N rows
  if (i >= GG * NN) return;
  const float* hp = h + (size_t)i * AT;
  float s = b2[0];
#pragma unroll 8
  for (int k = 0; k < AT; ++k) s += hp[k] * W2[k];
  logits[i] = s;
}

// ----------- softmax over graphs + weighted sum + residual + relu ----------

__global__ __launch_bounds__(256) void fuse_kernel(const float* __restrict__ feats,
                                                   const float* __restrict__ logits,
                                                   const float* __restrict__ resid,
                                                   float* __restrict__ out) {
  long long i = (long long)blockIdx.x * 256 + threadIdx.x;  // over N*D
  if (i >= (long long)NN * DD) return;
  const int n = (int)(i / DD);
  const float l0 = logits[n], l1 = logits[NN + n];
  const float l2 = logits[2 * NN + n], l3 = logits[3 * NN + n];
  const float mx = fmaxf(fmaxf(l0, l1), fmaxf(l2, l3));
  const float e0 = expf(l0 - mx), e1 = expf(l1 - mx);
  const float e2 = expf(l2 - mx), e3 = expf(l3 - mx);
  const float inv = 1.0f / (e0 + e1 + e2 + e3);
  const long long gs = (long long)NN * DD;
  float v = (e0 * feats[i] + e1 * feats[gs + i] + e2 * feats[2 * gs + i] +
             e3 * feats[3 * gs + i]) * inv;
  v += resid[i];
  out[i] = fmaxf(v, 0.0f);
}

// ---------- fusion_expression + passthrough copies of C matrices -----------

__global__ __launch_bounds__(256) void cmix_kernel(const float* __restrict__ cc,
                                                   const float* __restrict__ cs,
                                                   const float* __restrict__ fw,
                                                   float* __restrict__ out_fuse,
                                                   float* __restrict__ out_cc,
                                                   float* __restrict__ out_cs,
                                                   int n4) {
  int i = blockIdx.x * 256 + threadIdx.x;
  if (i >= n4) return;
  const float w0 = fw[0], w1 = fw[1];
  const float m = fmaxf(w0, w1);
  const float e0 = expf(w0 - m), e1 = expf(w1 - m);
  const float a = e0 / (e0 + e1), b = e1 / (e0 + e1);
  const float4 c0 = ((const float4*)cc)[i];
  const float4 c1 = ((const float4*)cs)[i];
  float4 f;
  f.x = a * c0.x + b * c1.x;
  f.y = a * c0.y + b * c1.y;
  f.z = a * c0.z + b * c1.z;
  f.w = a * c0.w + b * c1.w;
  ((float4*)out_fuse)[i] = f;
  ((float4*)out_cc)[i] = c0;
  ((float4*)out_cs)[i] = c1;
}

}  // namespace

// ---------------------------------------------------------------------------

extern "C" void kernel_launch(void* const* d_in, const int* in_sizes, int n_in,
                              void* d_out, int out_size, void* d_ws, size_t ws_size,
                              hipStream_t stream) {
  (void)in_sizes; (void)n_in; (void)out_size; (void)ws_size;

  const float* xs       = (const float*)d_in[0];   // [G,N,F]
  const int*   ei       = (const int*)d_in[1];     // [G,2,E]
  const float* xcontent = (const float*)d_in[2];   // [N,F]
  const float* sage_Wl  = (const float*)d_in[3];   // [G,3,H,H]
  const float* sage_bl  = (const float*)d_in[4];   // [G,3,H]
  const float* sage_Wr  = (const float*)d_in[5];   // [G,3,H,H]
  const float* lin_W    = (const float*)d_in[6];   // [G,D,F]
  const float* lin_b    = (const float*)d_in[7];   // [G,D]
  const float* att_W1   = (const float*)d_in[8];   // [ATT,D]
  const float* att_b1   = (const float*)d_in[9];   // [ATT]
  const float* att_W2   = (const float*)d_in[10];  // [1,ATT]
  const float* att_b2   = (const float*)d_in[11];  // [1]
  const float* res_W    = (const float*)d_in[12];  // [D,D]
  const float* res_b    = (const float*)d_in[13];  // [D]
  const float* fw       = (const float*)d_in[14];  // [2]
  const float* Cc       = (const float*)d_in[15];  // [N,N]
  const float* Cs       = (const float*)d_in[16];  // [N,N]

  float* out      = (float*)d_out;
  float* out_fuse = out;                               // N*N
  float* out_xc   = out_fuse + (size_t)NN * NN;        // N*F
  float* out_sf   = out_xc + (size_t)NN * FF;          // N*D
  float* out_cc   = out_sf + (size_t)NN * DD;          // N*N
  float* out_cs   = out_cc + (size_t)NN * NN;          // N*N

  // Workspace layout (floats)
  float* ws = (float*)d_ws;
  size_t off = 0;
  float* deg   = ws + off; off += (size_t)GG * NN;       // inverse degree
  float* xA    = ws + off; off += (size_t)GG * NN * HH;
  float* xB    = ws + off; off += (size_t)GG * NN * HH;
  float* magg  = ws + off; off += (size_t)GG * NN * HH;  // mean aggregation
  float* feats = ws + off; off += (size_t)GG * NN * DD;
  float* hbuf  = ws + off; off += (size_t)GG * NN * AT;
  float* lgts  = ws + off; off += (size_t)GG * NN;
  float* resid = ws + off; off += (size_t)NN * DD;

  // ---- degrees -> 1/max(deg,1) ----
  {
    int n4 = GG * NN / 4;
    fill_f32<<<(n4 + 255) / 256, 256, 0, stream>>>(deg, 0.0f, n4);
    count_deg<<<dim3((EE + 255) / 256, 1, GG), 256, 0, stream>>>(ei, deg);
    inv_deg<<<(GG * NN + 255) / 256, 256, 0, stream>>>(deg, GG * NN);
  }

  // ---- 3 GraphSAGE layers (all 4 graphs via grid.z) ----
  const float* xin = xs;
  float* xout = xA;
  for (int l = 0; l < 3; ++l) {
    int n4 = GG * NN * HH / 4;
    fill_f32<<<(n4 + 255) / 256, 256, 0, stream>>>(magg, 0.0f, n4);
    scatter_sum<<<dim3(EE / 4, 1, GG), 256, 0, stream>>>(xin, ei, magg);
    // relu( (m/deg) @ Wl^T + bl + x @ Wr^T )
    const int tiles = (NN / 16) * (HH / 64);  // 1536 -> 192 blocks
    wmma_gemm<<<dim3(tiles / 8, 1, GG), 256, 0, stream>>>(
        magg, deg, sage_Wl + (size_t)l * HH * HH,
        xin, sage_Wr + (size_t)l * HH * HH,
        sage_bl + (size_t)l * HH, xout,
        NN, HH, HH,
        (long long)NN * HH, (long long)3 * HH * HH, (long long)3 * HH,
        (long long)NN * HH, (long long)NN, /*relu=*/1);
    xin = xout;
    xout = (l == 0) ? xB : xA;
  }
  const float* xg = xin;  // final SAGE output [G,N,H] (== xA)

  // ---- feats = xg @ lin_W^T + lin_b  [G,N,D] ----
  {
    const int tiles = (NN / 16) * (DD / 64);  // 3072 -> 384 blocks
    wmma_gemm<<<dim3(tiles / 8, 1, GG), 256, 0, stream>>>(
        xg, nullptr, lin_W, nullptr, nullptr, lin_b, feats,
        NN, FF, DD,
        (long long)NN * HH, (long long)DD * FF, (long long)DD,
        (long long)NN * DD, 0, /*relu=*/0);
  }

  // ---- h = relu(feats @ att_W1^T + att_b1)  [G,N,ATT] (shared weights) ----
  {
    const int tiles = (NN / 16) * (AT / 64);  // 768 -> 96 blocks
    wmma_gemm<<<dim3(tiles / 8, 1, GG), 256, 0, stream>>>(
        feats, nullptr, att_W1, nullptr, nullptr, att_b1, hbuf,
        NN, DD, AT,
        (long long)NN * DD, 0, 0,
        (long long)NN * AT, 0, /*relu=*/1);
  }

  // ---- logits [G,N] ----
  logits_kernel<<<(GG * NN + 255) / 256, 256, 0, stream>>>(hbuf, att_W2, att_b2, lgts);

  // ---- residual = feats[0] @ res_W^T + res_b  [N,D] ----
  {
    const int tiles = (NN / 16) * (DD / 64);
    wmma_gemm<<<dim3(tiles / 8, 1, 1), 256, 0, stream>>>(
        feats, nullptr, res_W, nullptr, nullptr, res_b, resid,
        NN, DD, DD, 0, 0, 0, 0, 0, /*relu=*/0);
  }

  // ---- structure_features = relu(softmax-weighted feats + residual) ----
  fuse_kernel<<<(NN * DD + 255) / 256, 256, 0, stream>>>(feats, lgts, resid, out_sf);

  // ---- fusion_expression + C passthroughs (single N^2 pass) ----
  {
    int n4 = (int)((size_t)NN * NN / 4);
    cmix_kernel<<<(n4 + 255) / 256, 256, 0, stream>>>(Cc, Cs, fw, out_fuse,
                                                      out_cc, out_cs, n4);
  }

  // ---- x_content passthrough ----
  {
    int n4 = NN * FF / 4;
    copy_f32<<<(n4 + 255) / 256, 256, 0, stream>>>(out_xc, xcontent, n4);
  }
}
